// MLPDecoder_67912022884434
// MI455X (gfx1250) — matrix operand order repaired and checked by reading, not compiled, not run on hardware
//
#include <hip/hip_runtime.h>
#include <hip/hip_bf16.h>

// ---------------------------------------------------------------------------
// MLP edge-decoder (NRI-style message passing), fused for gfx1250.
//   GEMM1: adj(16x256) x W1^T(256x128)  -> bf16 WMMA, f32 acc  (x3 edge types)
//   GEMM2: h(16x128)  x W2^T(128x64)    -> bf16 WMMA, f32 acc  (x3 edge types)
//   adjacency gathered on the fly; triangular index computed incrementally.
//   weights pre-converted to bf16 once; blocks stage them with b128 copies.
// ---------------------------------------------------------------------------

typedef __attribute__((ext_vector_type(16))) __bf16 v16bf;
typedef __attribute__((ext_vector_type(8)))  float  v8f;

union Frag16 {
    unsigned int u[8];
    uint4        q[2];
    v16bf        v;
};

__device__ __forceinline__ unsigned short f32_to_bf16_rne(float f) {
    unsigned int u = __builtin_bit_cast(unsigned int, f);
    u += 0x7FFFu + ((u >> 16) & 1u);
    return (unsigned short)(u >> 16);
}

__device__ __forceinline__ unsigned int pack2_bf16(float a, float b) {
#if defined(__has_builtin)
#if __has_builtin(__builtin_amdgcn_cvt_pk_bf16_f32)
    typedef __attribute__((ext_vector_type(2))) __bf16 v2bf;
    v2bf r = __builtin_amdgcn_cvt_pk_bf16_f32(a, b);
    return __builtin_bit_cast(unsigned int, r);
#else
    return (unsigned int)f32_to_bf16_rne(a) | ((unsigned int)f32_to_bf16_rne(b) << 16);
#endif
#else
    return (unsigned int)f32_to_bf16_rne(a) | ((unsigned int)f32_to_bf16_rne(b) << 16);
#endif
}

#define NATOMS   256
#define NEDGES   32640
#define MSG_OUT  64

// LDS strides in bf16 elements; row byte sizes are multiples of 16 (b128-able)
// and dword strides (132/68/68/20) step lane start banks by 4 -> conflict-free.
#define W1_STRIDE 264   // 128 rows x 256 K
#define W2_STRIDE 136   // 64 rows x 128 K
#define H_STRIDE  136   // 16 rows x 128 K  (x3 edge types)
#define A_STRIDE  40    // 16 rows x 32 K   (x3 edge types, x2 buffers)

#define LDS_W1  (128 * W1_STRIDE)
#define LDS_W2  (64 * W2_STRIDE)
#define LDS_H   (3 * 16 * H_STRIDE)
#define LDS_A   (2 * 3 * 16 * A_STRIDE)
#define LDS_TOTAL_USHORT (LDS_W1 + LDS_W2 + LDS_H + LDS_A)

// workspace layout (bytes)
#define WS_OUTBUF_BYTES (NATOMS * NATOMS * MSG_OUT / NATOMS * 256 * 4)  // see launch
// (computed explicitly in kernel_launch)

// ---------------------------------------------------------------------------
// one-time: convert W1 (32768) and W2 (8192) f32 -> packed bf16 in workspace
// ---------------------------------------------------------------------------
__launch_bounds__(256)
__global__ void prep_weights_kernel(const float* __restrict__ w1,
                                    const float* __restrict__ w2,
                                    unsigned short* __restrict__ w1bf,
                                    unsigned short* __restrict__ w2bf)
{
    const int t = blockIdx.x * 256 + threadIdx.x;   // 0 .. 40959
    if (t < 128 * 256) {
        w1bf[t] = f32_to_bf16_rne(w1[t]);
    } else {
        const int u = t - 128 * 256;                // 0 .. 8191
        w2bf[u] = f32_to_bf16_rne(w2[u]);
    }
}

__launch_bounds__(128)
__global__ void msg_pass_kernel(const float* __restrict__ se,            // [B, 32640, 4]
                                const unsigned short* __restrict__ w1bf, // [128*256] bf16
                                const unsigned short* __restrict__ w2bf, // [64*128] bf16
                                const float* __restrict__ b1,            // [128]
                                const float* __restrict__ b2,            // [64]
                                float* __restrict__ outbuf)              // [B, 256, 64]
{
    extern __shared__ unsigned short smem[];
    unsigned short* w1s = smem;                  // bf16 [128][264]
    unsigned short* w2s = w1s + LDS_W1;          // bf16 [64][136]
    unsigned short* hs  = w2s + LDS_W2;          // bf16 [3][16][136]
    unsigned short* as_ = hs  + LDS_H;           // bf16 [2][3][16][40]

    const int tid   = threadIdx.x;               // 0..127
    const int wave  = tid >> 5;                  // 0..3
    const int lane  = tid & 31;
    const int nl    = lane & 15;
    const int khalf = lane >> 4;                 // 0 or 1

    const int bidx = blockIdx.x;
    const int b    = bidx >> 4;                  // batch
    const int i0   = (bidx & 15) << 4;           // 16-row tile base

    // ---- stage pre-converted bf16 weights into LDS with b128 copies ----
    {
        const uint4* src1 = (const uint4*)w1bf;  // 4096 uint4 chunks (8 bf16 each)
#pragma unroll
        for (int it = 0; it < 32; ++it) {
            const int c   = tid + it * 128;      // chunk id
            const int row = c >> 5;              // (c*8)/256
            const int col = (c & 31) * 8;
            *(uint4*)(w1s + row * W1_STRIDE + col) = src1[c];
        }
        const uint4* src2 = (const uint4*)w2bf;  // 1024 uint4 chunks
#pragma unroll
        for (int it = 0; it < 8; ++it) {
            const int c   = tid + it * 128;
            const int row = c >> 4;              // (c*8)/128
            const int col = (c & 15) * 8;
            *(uint4*)(w2s + row * W2_STRIDE + col) = src2[c];
        }
    }
    __syncthreads();

    // per-lane biases for this wave's output columns
    const float bias1a = b1[wave * 32 + nl];
    const float bias1b = b1[wave * 32 + 16 + nl];
    const float bias2  = b2[wave * 16 + nl];

    // GEMM1 accumulators: [edge type][n-tile 0/1]
    v8f c1[3][2] = {};

    const float* sbase = se + ((size_t)b * NEDGES) * 4;

    // ---------------- GEMM1: 8 k-steps of K=32, all 3 edge types ----------------
    for (int ks = 0; ks < 8; ++ks) {
        const int k0 = ks << 5;
        unsigned short* ab = as_ + (ks & 1) * (3 * 16 * A_STRIDE);

        // cooperative gather of 16x32 symmetric adjacency tile for ALL 3 channels.
        // One float4 load per edge element gives channels 1..3 at once; the
        // triangular index for 4 consecutive j is computed incrementally.
        {
            const int le = tid << 2;
            const int il = le >> 5;           // local row 0..15
            const int kk = le & 31;           // col within tile (multiple of 4)
            const int i  = i0 + il;
            const int j0 = k0 + kk;
            float4 ev[4];

            if (j0 > i) {
                // all four j > i: edge indices are consecutive
                const int idx = i * (511 - i) / 2 + (j0 - i - 1);
                const float* p = sbase + (size_t)idx * 4;
#pragma unroll
                for (int q = 0; q < 4; ++q) ev[q] = *(const float4*)(p + q * 4);
            } else if (j0 + 3 < i) {
                // all four j < i: idx(j+1) = idx(j) + (254 - j)
                const int idx = j0 * (511 - j0) / 2 + (i - j0 - 1);
                const int st  = 254 - j0;
                const float* p = sbase + (size_t)idx * 4;
                ev[0] = *(const float4*)(p);
                p += (size_t)st * 4;       ev[1] = *(const float4*)(p);
                p += (size_t)(st - 1) * 4; ev[2] = *(const float4*)(p);
                p += (size_t)(st - 2) * 4; ev[3] = *(const float4*)(p);
            } else {
                // span contains the diagonal: per-element
#pragma unroll
                for (int q = 0; q < 4; ++q) {
                    const int j = j0 + q;
                    if (j == i) {
                        ev[q] = make_float4(0.0f, 0.0f, 0.0f, 0.0f);
                    } else {
                        const int r  = (i < j) ? i : j;
                        const int c  = (i < j) ? j : i;
                        const int ix = r * (511 - r) / 2 + (c - r - 1);
                        ev[q] = *(const float4*)(sbase + (size_t)ix * 4);
                    }
                }
            }

            const int aoff = il * A_STRIDE + kk;
            *(unsigned long long*)(ab + aoff) =
                ((unsigned long long)pack2_bf16(ev[2].y, ev[3].y) << 32) | pack2_bf16(ev[0].y, ev[1].y);
            *(unsigned long long*)(ab + 16 * A_STRIDE + aoff) =
                ((unsigned long long)pack2_bf16(ev[2].z, ev[3].z) << 32) | pack2_bf16(ev[0].z, ev[1].z);
            *(unsigned long long*)(ab + 32 * A_STRIDE + aoff) =
                ((unsigned long long)pack2_bf16(ev[2].w, ev[3].w) << 32) | pack2_bf16(ev[0].w, ev[1].w);
        }
        __syncthreads();

        // B fragments from W1 (shared across the 3 edge types):
        // per lane K = khalf*8..+7 and 16+khalf*8..+7 = 2 x ds_load_b128
        Frag16 bfA, bfB;
        {
            const unsigned short* pa = w1s + (wave * 32 + nl) * W1_STRIDE + k0 + khalf * 8;
            const unsigned short* pb = pa + 16 * W1_STRIDE;
            bfA.q[0] = *(const uint4*)(pa);
            bfA.q[1] = *(const uint4*)(pa + 16);
            bfB.q[0] = *(const uint4*)(pb);
            bfB.q[1] = *(const uint4*)(pb + 16);
        }

#pragma unroll
        for (int e = 0; e < 3; ++e) {
            Frag16 a;
            const unsigned short* pr = ab + e * 16 * A_STRIDE + nl * A_STRIDE + khalf * 8;
            a.q[0] = *(const uint4*)(pr);        // K = khalf*8 .. +7
            a.q[1] = *(const uint4*)(pr + 16);   // K = 16+khalf*8 .. +7
            c1[e][0] = __builtin_amdgcn_wmma_f32_16x16x32_bf16(false, a.v, false, bfA.v,
                                                               (short)0, c1[e][0], false, false);
            c1[e][1] = __builtin_amdgcn_wmma_f32_16x16x32_bf16(false, a.v, false, bfB.v,
                                                               (short)0, c1[e][1], false, false);
        }
    }

    // ---- bias + relu + stage h[e] (16x128 bf16 each) in LDS ----
#pragma unroll
    for (int e = 0; e < 3; ++e) {
        unsigned short* he = hs + e * 16 * H_STRIDE;
#pragma unroll
        for (int v = 0; v < 8; ++v) {
            const int M = v + khalf * 8;          // C/D layout: M = v (+8 for hi lanes)
            const float ha = fmaxf(c1[e][0][v] + bias1a, 0.0f);
            const float hb = fmaxf(c1[e][1][v] + bias1b, 0.0f);
            he[M * H_STRIDE + wave * 32 + nl]      = f32_to_bf16_rne(ha);
            he[M * H_STRIDE + wave * 32 + 16 + nl] = f32_to_bf16_rne(hb);
        }
    }
    __syncthreads();

    // ---------------- GEMM2: 4 k-steps of K=32; W2 frag shared over edge types ----
    v8f c2[3] = {};
#pragma unroll
    for (int ks2 = 0; ks2 < 4; ++ks2) {
        const int k0 = ks2 << 5;
        Frag16 bw;
        {
            const unsigned short* pw = w2s + (wave * 16 + nl) * W2_STRIDE + k0 + khalf * 8;
            bw.q[0] = *(const uint4*)(pw);
            bw.q[1] = *(const uint4*)(pw + 16);
        }
#pragma unroll
        for (int e = 0; e < 3; ++e) {
            Frag16 a;
            const unsigned short* ph = hs + e * 16 * H_STRIDE + nl * H_STRIDE + k0 + khalf * 8;
            a.q[0] = *(const uint4*)(ph);
            a.q[1] = *(const uint4*)(ph + 16);
            c2[e] = __builtin_amdgcn_wmma_f32_16x16x32_bf16(false, a.v, false, bw.v,
                                                            (short)0, c2[e], false, false);
        }
    }

    // weighted sum over edge types: wts = [0.2, 0.4, 0.6]
    v8f out_acc = {};
#pragma unroll
    for (int e = 0; e < 3; ++e) {
        const float wt = 0.2f * (float)(e + 1);
#pragma unroll
        for (int v = 0; v < 8; ++v) {
            out_acc[v] = fmaf(wt, fmaxf(c2[e][v] + bias2, 0.0f), out_acc[v]);
        }
    }

    // ---- store out[b, i0+M, wave*16 + nl] ----
#pragma unroll
    for (int v = 0; v < 8; ++v) {
        const int M = v + khalf * 8;
        outbuf[((size_t)(b * NATOMS + i0 + M)) * MSG_OUT + wave * 16 + nl] = out_acc[v];
    }
}

// ---------------------------------------------------------------------------
// Head: p = relu(out@wo1^T+bo1); p = relu(p@wo2^T+bo2); mean over atoms;
//       result = p_mean @ wo3^T + bo3.   One block per batch, thread = atom row.
// ---------------------------------------------------------------------------
__launch_bounds__(256)
__global__ void head_kernel(const float* __restrict__ outbuf, // [B,256,64]
                            const float* __restrict__ wo1,    // [64,64]
                            const float* __restrict__ bo1,    // [64]
                            const float* __restrict__ wo2,    // [64,64]
                            const float* __restrict__ bo2,    // [64]
                            const float* __restrict__ wo3,    // [2,64]
                            const float* __restrict__ bo3,    // [2]
                            float* __restrict__ out)          // [B,2]
{
    __shared__ float s_wo1[64 * 64];
    __shared__ float s_wo2[64 * 64];
    __shared__ float s_mean[64];

    const int tid = threadIdx.x;      // atom row 0..255
    const int b   = blockIdx.x;

    for (int i = tid; i < 64 * 64; i += 256) {
        s_wo1[i] = wo1[i];
        s_wo2[i] = wo2[i];
    }
    if (tid < 64) s_mean[tid] = 0.0f;
    __syncthreads();

    float x[64], p[64];
    const float* row = outbuf + ((size_t)b * NATOMS + tid) * MSG_OUT;
#pragma unroll
    for (int k = 0; k < 64; ++k) x[k] = row[k];

#pragma unroll 4
    for (int j = 0; j < 64; ++j) {
        float acc = bo1[j];
        const float* wr = s_wo1 + j * 64;
#pragma unroll
        for (int k = 0; k < 64; ++k) acc = fmaf(x[k], wr[k], acc);
        p[j] = fmaxf(acc, 0.0f);
    }
#pragma unroll 4
    for (int j = 0; j < 64; ++j) {
        float acc = bo2[j];
        const float* wr = s_wo2 + j * 64;
#pragma unroll
        for (int k = 0; k < 64; ++k) acc = fmaf(p[k], wr[k], acc);
        atomicAdd(&s_mean[j], fmaxf(acc, 0.0f));
    }
    __syncthreads();

    if (tid < 2) {
        float acc = bo3[tid];
        const float* wr = wo3 + tid * 64;
#pragma unroll
        for (int k = 0; k < 64; ++k) acc = fmaf(s_mean[k] * (1.0f / 256.0f), wr[k], acc);
        out[b * 2 + tid] = acc;
    }
}

// ---------------------------------------------------------------------------
extern "C" void kernel_launch(void* const* d_in, const int* in_sizes, int n_in,
                              void* d_out, int out_size, void* d_ws, size_t ws_size,
                              hipStream_t stream) {
    // setup_inputs order:
    // 0 inputs (UNUSED by reference), 1 sparse_edges, 2 w1, 3 b1, 4 w2, 5 b2,
    // 6 wo1, 7 bo1, 8 wo2, 9 bo2, 10 wo3, 11 bo3
    const float* se  = (const float*)d_in[1];
    const float* w1  = (const float*)d_in[2];
    const float* b1  = (const float*)d_in[3];
    const float* w2  = (const float*)d_in[4];
    const float* b2  = (const float*)d_in[5];
    const float* wo1 = (const float*)d_in[6];
    const float* bo1 = (const float*)d_in[7];
    const float* wo2 = (const float*)d_in[8];
    const float* bo2 = (const float*)d_in[9];
    const float* wo3 = (const float*)d_in[10];
    const float* bo3 = (const float*)d_in[11];

    // workspace: [outbuf f32 16 MB][w1 bf16 64 KB][w2 bf16 16 KB]
    char* ws = (char*)d_ws;
    float*          outbuf = (float*)ws;                                   // 256*256*64 f32
    unsigned short* w1bf   = (unsigned short*)(ws + (size_t)256 * 256 * 64 * 4);
    unsigned short* w2bf   = w1bf + 128 * 256;

    prep_weights_kernel<<<dim3(160), dim3(256), 0, stream>>>(w1, w2, w1bf, w2bf);

    const size_t shmem = (size_t)LDS_TOTAL_USHORT * sizeof(unsigned short); // ~103 KB

    // 256 batches x 16 row-tiles, 4 waves per block
    msg_pass_kernel<<<dim3(256 * 16), dim3(128), shmem, stream>>>(
        se, w1bf, w2bf, b1, b2, outbuf);

    head_kernel<<<dim3(256), dim3(256), 0, stream>>>(
        outbuf, wo1, bo1, wo2, bo2, wo3, bo3, (float*)d_out);
}
